// FractionalFusionUp_11991548690812
// MI455X (gfx1250) — matrix operand order repaired and checked by reading, not compiled, not run on hardware
//
#include <hip/hip_runtime.h>
#include <hip/hip_bf16.h>
#include <math.h>

// ---------------------------------------------------------------------------
// FractionalFusionUp on MI455X (gfx1250, wave32)
//   conv1 (21->96 fused heads) and conv2 (96->9 fused heads) as implicit-GEMM
//   on v_wmma_f32_16x16x32_bf16 with weights staged into LDS by the Tensor
//   Data Mover (tensor_load_to_lds + s_wait_tensorcnt).  Fully branchless
//   im2col via an LDS K-lookup table + per-lane tap-validity bitmask.
//   Per-pixel GL epilogue at low res, then a bandwidth-bound HR combine
//   (output ~38MB -> memory-bound at 23.3TB/s).
// ---------------------------------------------------------------------------

typedef __attribute__((ext_vector_type(16))) __bf16 v16bf;
typedef __attribute__((ext_vector_type(8)))  float  v8f;
typedef unsigned int ui4 __attribute__((ext_vector_type(4)));
typedef int          si8 __attribute__((ext_vector_type(8)));
typedef int          si4 __attribute__((ext_vector_type(4)));

#define B_     4
#define C_IN   21
#define H0     81
#define W0     81
#define HH     324            // H0*4
#define NCH    96             // fused hidden channels (48 orient + 48 gate)
#define NPIX   (B_*H0*W0)     // 26244 low-res pixels
#define NTILE  ((NPIX+15)/16) // 1641 wave tiles of 16 pixels
#define K1     189            // 21*9 im2col depth (conv1)
#define K1P    192
#define KC1    6              // K chunks of 32
#define NT1    6              // 96/16 N tiles
#define K2     864            // 96*9 im2col depth (conv2)
#define K2P    896
#define KC2    28
#define NHR    (B_*HH*HH)     // 419904 high-res pixels

#define W1ELEMS (KC1*NT1*32*16)   // 18432 bf16 = 36864 B = 9216 dwords
#define W2ELEMS (KC2*32*16)       // 14336 bf16 = 28672 B = 7168 dwords

// workspace layout (bytes, 256-aligned)
#define OFF_W1  0u
#define OFF_W2  36864u
#define OFF_H   65536u            // hidden bf16 [B][96][81][81] = 5038848 B
#define OFF_MLP 5104384u          // mask_lp f32 [NPIX][9]
#define OFF_HP  6049280u          // ker_hp  f32 [NPIX][9]
#define OFF_G   6994176u          // g f32 [NPIX]

// output layout (floats): (y, v, g_up) concatenated
#define OUT_V   (B_*C_IN*HH*HH)        // 8817984
#define OUT_G   (OUT_V + B_*8*H0*W0)   // 9027936

// signed step table s_d[kl]: ker[kl] = -1/8 * sum_d v[d]*S8[d][kl]
__device__ __constant__ float S8[8][9] = {
  {-1, 0, 1,  -1, 0, 1,  -1, 0, 1},   // ( 1, 0): sign(dx)
  {-1,-1, 0,  -1, 0, 1,   0, 1, 1},   // ( 1, 1): sign(dy+dx)
  {-1,-1,-1,   0, 0, 0,   1, 1, 1},   // ( 0, 1): sign(dy)
  { 0,-1,-1,   1, 0,-1,   1, 1, 0},   // (-1, 1): sign(dy-dx)
  { 1, 0,-1,   1, 0,-1,   1, 0,-1},   // (-1, 0): sign(-dx)
  { 1, 1, 0,   1, 0,-1,   0,-1,-1},   // (-1,-1): -sign(dy+dx)
  { 1, 1, 1,   0, 0, 0,  -1,-1,-1},   // ( 0,-1): sign(-dy)
  { 0, 1, 1,  -1, 0, 1,  -1,-1, 0},   // ( 1,-1): sign(dx-dy)
};

// A-fragment (16x32 bf16) within-lane K map
__device__ __forceinline__ int a_kmap(int e, int half) {
  return e + 16 * half;
}

// per-lane 9-bit tap-validity mask (bit t = tap (ty,tx) in-bounds for pixel)
__device__ __forceinline__ int tap_mask(int py, int px) {
  int tm = 0;
#pragma unroll
  for (int i = 0; i < 3; ++i)
#pragma unroll
    for (int j = 0; j < 3; ++j) {
      int okt = (int)((unsigned)(py + i - 1) < (unsigned)H0) &
                (int)((unsigned)(px + j - 1) < (unsigned)W0);
      tm |= okt << (i * 3 + j);
    }
  return tm;
}

// ---------------------------------------------------------------------------
// TDM: 1-D descriptor copy of `ndwords` dwords global -> LDS.
// ---------------------------------------------------------------------------
__device__ __forceinline__ void tdm_copy_to_lds(const void* gsrc, void* ldst,
                                                unsigned ndwords) {
  unsigned long long ga = (unsigned long long)(uintptr_t)gsrc;
  unsigned lo = (unsigned)(uintptr_t)ldst;     // low 32 bits = LDS offset
  ui4 g0;
  g0[0] = 1u;                                  // count=1, user descriptor
  g0[1] = lo;                                  // lds_addr
  g0[2] = (unsigned)ga;                        // global_addr[31:0]
  g0[3] = (unsigned)(ga >> 32) | 0x80000000u;  // global_addr[56:32] | type=2
  si8 g1;
  g1[0] = 0x20000;                             // data_size=2 (4 bytes)
  g1[1] = (int)((ndwords & 0xFFFFu) << 16);    // tensor_dim0[15:0]
  g1[2] = (int)((ndwords >> 16) & 0xFFFFu) | (1 << 16);  // td0 hi | tensor_dim1=1
  g1[3] = (int)((ndwords & 0xFFFFu) << 16);    // tile_dim0
  g1[4] = 0;                                   // tile_dim1=0 (1-D), tile_dim2=0
  g1[5] = (int)ndwords;                        // tensor_dim0_stride[31:0]
  g1[6] = 0;
  g1[7] = 0;
  si4 z4 = {0, 0, 0, 0};
#if defined(__clang_major__) && (__clang_major__ >= 23)
  si8 z8 = {0, 0, 0, 0, 0, 0, 0, 0};
  __builtin_amdgcn_tensor_load_to_lds(g0, g1, z4, z4, z8, 0);
#else
  __builtin_amdgcn_tensor_load_to_lds(g0, g1, z4, z4, 0);
#endif
}

// ---------------------------------------------------------------------------
// Weight packing into exact B-fragment order.
// ---------------------------------------------------------------------------
__global__ void prep_w1(const float* __restrict__ w1o,
                        const float* __restrict__ w1g,
                        __bf16* __restrict__ wp) {
  int t = blockIdx.x * blockDim.x + threadIdx.x;
  if (t >= W1ELEMS) return;
  int e  = t & 15;
  int l  = (t >> 4) & 31;
  int rs = t >> 9;
  int tn = rs % NT1;
  int kc = rs / NT1;
  int half = l >> 4, lq = l & 15;
  int K = kc * 32 + e + 16 * half;
  int n = tn * 16 + lq;
  float v = 0.f;
  if (K < K1) {
    int ci = K / 9, tt = K - ci * 9, ty = tt / 3, tx = tt - ty * 3;
    v = (n < 48) ? w1o[((n * C_IN + ci) * 3 + ty) * 3 + tx]
                 : w1g[(((n - 48) * C_IN + ci) * 3 + ty) * 3 + tx];
  }
  wp[t] = (__bf16)v;
}

__global__ void prep_w2(const float* __restrict__ w2o,
                        const float* __restrict__ w2g,
                        __bf16* __restrict__ wp) {
  int t = blockIdx.x * blockDim.x + threadIdx.x;
  if (t >= W2ELEMS) return;
  int e  = t & 15;
  int l  = (t >> 4) & 31;
  int kc = t >> 9;
  int half = l >> 4, lq = l & 15;
  int K = kc * 32 + e + 16 * half;
  int n = lq;
  float v = 0.f;
  if (K < K2) {
    int ci = K / 9, tt = K - ci * 9, ty = tt / 3, tx = tt - ty * 3;
    if (n < 8 && ci < 48)        v = w2o[((n * 48 + ci) * 3 + ty) * 3 + tx];
    else if (n == 8 && ci >= 48) v = w2g[((ci - 48) * 3 + ty) * 3 + tx];
  }
  wp[t] = (__bf16)v;
}

// K-table entry: (addr_off << 5) | (valid << 4) | tap   (tap = 0..8)
__device__ __forceinline__ int make_kent(int K, int Kmax, int chanStride) {
  if (K >= Kmax) return 0;                     // invalid: off=0, valid=0, tap=0
  int ci = K / 9, tap = K - ci * 9, ty = tap / 3, tx = tap - ty * 3;
  int off = ci * chanStride + (ty - 1) * W0 + (tx - 1);
  return (off << 5) | (1 << 4) | tap;
}

// ---------------------------------------------------------------------------
// conv1: x[B,21,81,81] -> relu -> hidden bf16 [B,96,81,81]   (both heads)
// ---------------------------------------------------------------------------
__global__ void __launch_bounds__(256)
conv1_wmma(const float* __restrict__ x,
           const float* __restrict__ b1o, const float* __restrict__ b1g,
           const __bf16* __restrict__ wp, __bf16* __restrict__ hbuf) {
  __shared__ __attribute__((aligned(32))) __bf16 w1s[W1ELEMS];
  __shared__ int ktab[K1P];

  for (int i = threadIdx.x; i < K1P; i += blockDim.x)
    ktab[i] = make_kent(i, K1, H0 * W0);
  if ((threadIdx.x >> 5) == 0) {               // TDM weight staging, wave 0
    tdm_copy_to_lds(wp, w1s, W1ELEMS / 2);
    __builtin_amdgcn_s_wait_tensorcnt(0);
  }
  __syncthreads();

  int wave = blockIdx.x * 8 + (threadIdx.x >> 5);
  if (wave >= NTILE) wave = NTILE - 1;         // duplicate work, same stores
  int lane = threadIdx.x & 31;
  int half = lane >> 4, lq = lane & 15;
  int p0 = wave * 16;
  int pr = p0 + lq; if (pr >= NPIX) pr = NPIX - 1;
  int b   = pr / (H0 * W0);
  int rem = pr - b * H0 * W0;
  int py  = rem / W0, px = rem - py * W0;
  int pixbase = b * (C_IN * H0 * W0) + py * W0 + px;
  int tmask = tap_mask(py, px);

  v8f acc[NT1] = {};
  const v16bf* wls = (const v16bf*)w1s;

  for (int kc = 0; kc < KC1; ++kc) {
    v16bf a;
#pragma unroll
    for (int e = 0; e < 16; ++e) {
      int K = kc * 32 + a_kmap(e, half);
      int ent = ktab[K];
      int ok = (ent >> 4) & (tmask >> (ent & 15)) & 1;   // branchless predicate
      int off = ok ? (ent >> 5) : 0;                     // always-safe address
      float v = x[pixbase + off];
      a[e] = (__bf16)(ok ? v : 0.f);
    }
#pragma unroll
    for (int tn = 0; tn < NT1; ++tn) {
      v16bf bf = wls[(kc * NT1 + tn) * 32 + lane];
      acc[tn] = __builtin_amdgcn_wmma_f32_16x16x32_bf16(
          false, a, false, bf, (short)0, acc[tn], false, false);
    }
  }

  // store: D rows M = r + 8*half -> pixels p0+8*half+r (consecutive)
  int sbase[8];
  bool sok[8];
  {
    int pix = p0 + 8 * half;
    int bb = pix / (H0 * W0);
    int rr = pix - bb * (H0 * W0);
    int yy = rr / W0, xx = rr - yy * W0;
#pragma unroll
    for (int r = 0; r < 8; ++r) {
      sok[r]   = (pix < NPIX);
      sbase[r] = ((bb * NCH) * H0 + yy) * W0 + xx;
      ++pix; ++xx;
      if (xx == W0) { xx = 0; ++yy; if (yy == H0) { yy = 0; ++bb; } }
    }
  }
#pragma unroll
  for (int tn = 0; tn < NT1; ++tn) {
    int n = tn * 16 + lq;
    float bias = (n < 48) ? b1o[n] : b1g[n - 48];
#pragma unroll
    for (int r = 0; r < 8; ++r) {
      if (sok[r]) {
        float v = fmaxf(acc[tn][r] + bias, 0.f);
        hbuf[sbase[r] + n * (H0 * W0)] = (__bf16)v;
      }
    }
  }
}

// ---------------------------------------------------------------------------
// conv2 + GL epilogue: hidden -> (8 dirs + gate) via WMMA, then per-pixel
// tanh/sigmoid, ker, softmax mask, highpass mask.
// ---------------------------------------------------------------------------
__global__ void __launch_bounds__(256)
conv2_wmma(const __bf16* __restrict__ hbuf, const __bf16* __restrict__ wp,
           const float* __restrict__ b2o, const float* __restrict__ b2g,
           float* __restrict__ mask_lp, float* __restrict__ ker_hp,
           float* __restrict__ gbuf, float* __restrict__ vout) {
  __shared__ __attribute__((aligned(32))) __bf16 w2s[W2ELEMS];
  __shared__ int ktab2[K2P];
  __shared__ float s_raw[8][16][16];

  for (int i = threadIdx.x; i < K2P; i += blockDim.x)
    ktab2[i] = make_kent(i, K2, H0 * W0);
  if ((threadIdx.x >> 5) == 0) {
    tdm_copy_to_lds(wp, w2s, W2ELEMS / 2);
    __builtin_amdgcn_s_wait_tensorcnt(0);
  }
  __syncthreads();

  int widx = threadIdx.x >> 5;
  int wave = blockIdx.x * 8 + widx;
  if (wave >= NTILE) wave = NTILE - 1;
  int lane = threadIdx.x & 31;
  int half = lane >> 4, lq = lane & 15;
  int p0 = wave * 16;
  int pr = p0 + lq; if (pr >= NPIX) pr = NPIX - 1;
  int b   = pr / (H0 * W0);
  int rem = pr - b * H0 * W0;
  int py  = rem / W0, px = rem - py * W0;
  int pixbase = b * (NCH * H0 * W0) + py * W0 + px;
  int tmask = tap_mask(py, px);

  v8f acc = {};
  const v16bf* wls = (const v16bf*)w2s;

  for (int kc = 0; kc < KC2; ++kc) {
    v16bf a;
#pragma unroll
    for (int e = 0; e < 16; ++e) {
      int K = kc * 32 + a_kmap(e, half);
      int ent = ktab2[K];
      int ok = (ent >> 4) & (tmask >> (ent & 15)) & 1;
      int off = ok ? (ent >> 5) : 0;
      __bf16 v = hbuf[pixbase + off];
      a[e] = ok ? v : (__bf16)0.f;
    }
    v16bf bf = wls[kc * 32 + lane];
    acc = __builtin_amdgcn_wmma_f32_16x16x32_bf16(
        false, a, false, bf, (short)0, acc, false, false);
  }

  float bias = (lq < 8) ? b2o[lq] : ((lq == 8) ? b2g[0] : 0.f);
#pragma unroll
  for (int r = 0; r < 8; ++r)
    s_raw[widx][r + 8 * half][lq] = acc[r] + bias;
  __syncthreads();

  if (lane < 16) {
    int p = p0 + lane;
    if (p < NPIX) {
      float raw[9];
#pragma unroll
      for (int n = 0; n < 9; ++n) raw[n] = s_raw[widx][lane][n];

      float vv[8];
#pragma unroll
      for (int d = 0; d < 8; ++d) vv[d] = tanhf(raw[d]);   // VMAX = 1.0
      float g = 1.f / (1.f + expf(-raw[8]));

      int bb = p / (H0 * W0);
      int rr = p - bb * H0 * W0;
      int yy = rr / W0, xx = rr - yy * W0;
#pragma unroll
      for (int d = 0; d < 8; ++d)
        vout[((bb * 8 + d) * H0 + yy) * W0 + xx] = vv[d];
      gbuf[p] = g;

      float kern[9];
#pragma unroll
      for (int kl = 0; kl < 9; ++kl) {
        float s = 0.f;
#pragma unroll
        for (int d = 0; d < 8; ++d) s += vv[d] * S8[d][kl];
        kern[kl] = -0.125f * s;
      }
      kern[4] += 2.5f;                       // CENTER_BIAS (biases ker too)

      float mx = kern[0];
#pragma unroll
      for (int kl = 1; kl < 9; ++kl) mx = fmaxf(mx, kern[kl]);
      float ex[9], se = 0.f;
#pragma unroll
      for (int kl = 0; kl < 9; ++kl) {       // softmax(ker / TAU), TAU = 0.5
        ex[kl] = expf((kern[kl] - mx) * 2.0f);
        se += ex[kl];
      }
      float inv = 1.f / se;
#pragma unroll
      for (int kl = 0; kl < 9; ++kl) mask_lp[p * 9 + kl] = ex[kl] * inv;

      float mean = 0.f;
#pragma unroll
      for (int kl = 0; kl < 9; ++kl) mean += kern[kl];
      mean *= (1.f / 9.f);
      float tdev[9], den = 1e-8f;
#pragma unroll
      for (int kl = 0; kl < 9; ++kl) {
        tdev[kl] = kern[kl] - mean;
        den += fabsf(tdev[kl]);
      }
      float rden = 1.f / den;
#pragma unroll
      for (int kl = 0; kl < 9; ++kl) ker_hp[p * 9 + kl] = tdev[kl] * rden;
    }
  }
}

// ---------------------------------------------------------------------------
// HR combine (bandwidth-bound): nearest-up masks/g, bilinear x_up
// (align_corners), reflect-padded 9-tap gather, final blend.
// ---------------------------------------------------------------------------
__global__ void __launch_bounds__(256)
upsample_combine(const float* __restrict__ x,
                 const float* __restrict__ mask_lp,
                 const float* __restrict__ ker_hp,
                 const float* __restrict__ gbuf,
                 const float* __restrict__ beta_p,
                 float* __restrict__ yout, float* __restrict__ gout) {
  int tid = blockIdx.x * blockDim.x + threadIdx.x;
  if (tid >= NHR) return;
  int b   = tid / (HH * HH);
  int rem = tid - b * HH * HH;
  int Y = rem / HH, X = rem - Y * HH;
  int yl = Y >> 2, xl = X >> 2;
  int p = (b * H0 + yl) * W0 + xl;

  float m[9], hk[9];
#pragma unroll
  for (int n = 0; n < 9; ++n) { m[n] = mask_lp[p * 9 + n]; hk[n] = ker_hp[p * 9 + n]; }
  float g   = gbuf[p];
  float lam = 0.15f * (1.f - g);             // SMOOTH_RES
  float tb  = tanhf(*beta_p) * g;
  gout[tid] = g;

  int ry[3], rx[3];
#pragma unroll
  for (int i = 0; i < 3; ++i) {
    int t = yl + i - 1;
    ry[i] = (t < 0) ? -t : ((t > H0 - 1) ? 2 * (H0 - 1) - t : t);
    t = xl + i - 1;
    rx[i] = (t < 0) ? -t : ((t > W0 - 1) ? 2 * (W0 - 1) - t : t);
  }

  const float sc = (float)(H0 - 1) / (float)(HH - 1);
  float fy = Y * sc, fx = X * sc;
  int y0 = (int)fy; if (y0 > H0 - 2) y0 = H0 - 2;
  int x0 = (int)fx; if (x0 > W0 - 2) x0 = W0 - 2;
  float wy = fy - y0, wx = fx - x0;
  float w00 = (1.f - wy) * (1.f - wx), w01 = (1.f - wy) * wx;
  float w10 = wy * (1.f - wx),         w11 = wy * wx;

  for (int c = 0; c < C_IN; ++c) {
    const float* xb = x + (size_t)((b * C_IN + c) * H0) * W0;
    float nlp = 0.f, nhp = 0.f;
#pragma unroll
    for (int i = 0; i < 3; ++i)
#pragma unroll
      for (int j = 0; j < 3; ++j) {
        float xv = xb[ry[i] * W0 + rx[j]];
        int n = i * 3 + j;
        nlp += m[n] * xv;
        nhp += hk[n] * xv;
      }
    float xup = xb[y0 * W0 + x0] * w00 + xb[y0 * W0 + x0 + 1] * w01
              + xb[(y0 + 1) * W0 + x0] * w10 + xb[(y0 + 1) * W0 + x0 + 1] * w11;
    float out = xup + lam * (nlp - xup) + tb * nhp;
    yout[((size_t)(b * C_IN + c) * HH + Y) * HH + X] = out;
  }
}

// ---------------------------------------------------------------------------
extern "C" void kernel_launch(void* const* d_in, const int* in_sizes, int n_in,
                              void* d_out, int out_size, void* d_ws, size_t ws_size,
                              hipStream_t stream) {
  const float* x    = (const float*)d_in[0];
  const float* w1o  = (const float*)d_in[1];
  const float* b1o  = (const float*)d_in[2];
  const float* w2o  = (const float*)d_in[3];
  const float* b2o  = (const float*)d_in[4];
  const float* w1g  = (const float*)d_in[5];
  const float* b1g  = (const float*)d_in[6];
  const float* w2g  = (const float*)d_in[7];
  const float* b2g  = (const float*)d_in[8];
  const float* beta = (const float*)d_in[9];

  char* ws = (char*)d_ws;
  __bf16* wp1  = (__bf16*)(ws + OFF_W1);
  __bf16* wp2  = (__bf16*)(ws + OFF_W2);
  __bf16* hbuf = (__bf16*)(ws + OFF_H);
  float*  mlp  = (float*)(ws + OFF_MLP);
  float*  hpb  = (float*)(ws + OFF_HP);
  float*  gb   = (float*)(ws + OFF_G);

  float* out  = (float*)d_out;
  float* yout = out;
  float* vout = out + OUT_V;
  float* gout = out + OUT_G;

  prep_w1<<<(W1ELEMS + 255) / 256, 256, 0, stream>>>(w1o, w1g, wp1);
  prep_w2<<<(W2ELEMS + 255) / 256, 256, 0, stream>>>(w2o, w2g, wp2);
  conv1_wmma<<<(NTILE + 7) / 8, 256, 0, stream>>>(x, b1o, b1g, wp1, hbuf);
  conv2_wmma<<<(NTILE + 7) / 8, 256, 0, stream>>>(hbuf, wp2, b2o, b2g,
                                                  mlp, hpb, gb, vout);
  upsample_combine<<<(NHR + 255) / 256, 256, 0, stream>>>(x, mlp, hpb, gb,
                                                          beta, yout, gout);
}
